// SchNetCFconv_24953759989864
// MI455X (gfx1250) — matrix operand order repaired and checked by reading, not compiled, not run on hardware
//
#include <hip/hip_runtime.h>

#define F 128
#define WP 136          // LDS row pitch in bf16 elems: 272 bytes, multiple of 16B
#define WAVES 8
#define TILE_M 16
#define BLOCK_EDGES (WAVES * TILE_M)   // 128 edges per workgroup

typedef __attribute__((ext_vector_type(16))) __bf16 v16bf;
typedef __attribute__((ext_vector_type(8)))  __bf16 v8bf;
typedef __attribute__((ext_vector_type(8)))  float  v8f;
typedef __attribute__((ext_vector_type(4)))  float  v4f;

union bfu { v16bf v; v8bf h[2]; };

__device__ __forceinline__ float shifted_softplus(float x) {
    // softplus(x) - log(2), numerically stable
    return fmaxf(x, 0.0f) + log1pf(__expf(-fabsf(x))) - 0.69314718056f;
}

__global__ void __launch_bounds__(256, 1)
cfconv_zero_kernel(float* __restrict__ p, long long n) {
    long long i = (long long)blockIdx.x * blockDim.x + threadIdx.x;
    if (i < n) p[i] = 0.0f;
}

__global__ void __launch_bounds__(256, 1)
cfconv_fused_kernel(const float* __restrict__ nodes,
                    const float* __restrict__ edges,
                    const int*   __restrict__ edge_index,
                    const float* __restrict__ W1,
                    const float* __restrict__ b1,
                    const float* __restrict__ W2,
                    const float* __restrict__ b2,
                    float* __restrict__ out,
                    int M /* edges */, int Nn /* nodes */) {
    __shared__ __align__(16) __bf16 w1t[F][WP];            // W1 transposed: [n][k]
    __shared__ __align__(16) __bf16 w2t[F][WP];            // W2 transposed: [n][k]
    __shared__ __align__(16) __bf16 hsh[WAVES][TILE_M][WP]; // per-wave H strip
    __shared__ float b1s[F];
    __shared__ float b2s[F];

    const int tid = threadIdx.x;

    // ---- Stage weights (transposed, bf16) and biases into LDS ----
    #pragma unroll 4
    for (int i = tid; i < F * F; i += 256) {
        int k = i >> 7;
        int n = i & (F - 1);
        w1t[n][k] = (__bf16)W1[i];   // W1 row-major [k][n]
        w2t[n][k] = (__bf16)W2[i];
    }
    if (tid < F) { b1s[tid] = b1[tid]; b2s[tid] = b2[tid]; }
    __syncthreads();

    const int wave = tid >> 5;
    const int lane = tid & 31;
    const int l16  = lane & 15;
    const int hi   = lane >> 4;   // lane half (K-split selector per ISA layout)

    // A-frag source row for this lane (lanes l and l+16 share row M=l)
    const int e_a = blockIdx.x * BLOCK_EDGES + wave * TILE_M + l16;
    const float* arow = edges + (long long)(e_a < M ? e_a : (M - 1)) * F;

    // =================== GEMM1: H = ssp(E @ W1 + b1) ===================
    v8f acc[8];
    #pragma unroll
    for (int nb = 0; nb < 8; ++nb)
        #pragma unroll
        for (int j = 0; j < 8; ++j) acc[nb][j] = 0.0f;

    #pragma unroll
    for (int kk = 0; kk < 4; ++kk) {
        // A (16x32 bf16): lane half 0 -> K {k0..k0+7, k0+16..k0+23}
        //                 lane half 1 -> K {k0+8..k0+15, k0+24..k0+31}
        const int ka = kk * 32 + hi * 8;
        v4f a0 = *(const v4f*)(arow + ka);
        v4f a1 = *(const v4f*)(arow + ka + 4);
        v4f a2 = *(const v4f*)(arow + ka + 16);
        v4f a3 = *(const v4f*)(arow + ka + 20);
        bfu A;
        #pragma unroll
        for (int j = 0; j < 4; ++j) {
            A.v[j]      = (__bf16)a0[j];
            A.v[4 + j]  = (__bf16)a1[j];
            A.v[8 + j]  = (__bf16)a2[j];
            A.v[12 + j] = (__bf16)a3[j];
        }
        // B (32x16 bf16): lane half 0 -> K {k0..k0+7, k0+16..k0+23}, half 1 -> K {k0+8..k0+15, k0+24..k0+31}
        const int kb = kk * 32 + hi * 8;
        #pragma unroll
        for (int nb = 0; nb < 8; ++nb) {
            bfu B;
            B.h[0] = *(const v8bf*)&w1t[nb * 16 + l16][kb];
            B.h[1] = *(const v8bf*)&w1t[nb * 16 + l16][kb + 16];
            acc[nb] = __builtin_amdgcn_wmma_f32_16x16x32_bf16(
                false, A.v, false, B.v, (short)0, acc[nb], false, false);
        }
    }

    // bias + shifted softplus, spill H strip to LDS as bf16
    // C/D layout: lane<16 holds col N=lane, rows 0..7; lane>=16 rows 8..15
    #pragma unroll
    for (int nb = 0; nb < 8; ++nb) {
        const int n = nb * 16 + l16;
        const float bv = b1s[n];
        #pragma unroll
        for (int r = 0; r < 8; ++r) {
            float h = shifted_softplus(acc[nb][r] + bv);
            hsh[wave][hi * 8 + r][n] = (__bf16)h;
        }
    }
    __syncthreads();  // cross-lane LDS visibility within each wave strip

    // =================== GEMM2: X = H @ W2 + b2 ===================
    #pragma unroll
    for (int nb = 0; nb < 8; ++nb)
        #pragma unroll
        for (int j = 0; j < 8; ++j) acc[nb][j] = 0.0f;

    #pragma unroll
    for (int kk = 0; kk < 4; ++kk) {
        const int ka = kk * 32 + hi * 8;
        bfu A;
        A.h[0] = *(const v8bf*)&hsh[wave][l16][ka];
        A.h[1] = *(const v8bf*)&hsh[wave][l16][ka + 16];
        const int kb = kk * 32 + hi * 8;
        #pragma unroll
        for (int nb = 0; nb < 8; ++nb) {
            bfu B;
            B.h[0] = *(const v8bf*)&w2t[nb * 16 + l16][kb];
            B.h[1] = *(const v8bf*)&w2t[nb * 16 + l16][kb + 16];
            acc[nb] = __builtin_amdgcn_wmma_f32_16x16x32_bf16(
                false, A.v, false, B.v, (short)0, acc[nb], false, false);
        }
    }

    // =================== Epilogue: gather * X, scatter-add ===================
    const int base_e = blockIdx.x * BLOCK_EDGES + wave * TILE_M + hi * 8;
    int rcv[8], snd[8];
    #pragma unroll
    for (int r = 0; r < 8; ++r) {
        const int e  = base_e + r;
        const int ec = (e < M) ? e : 0;
        rcv[r] = (e < M) ? edge_index[ec] : -1;          // receiver (aggregation target)
        snd[r] = edge_index[(long long)M + ec];           // sender (gather source)
    }
    #pragma unroll
    for (int nb = 0; nb < 8; ++nb) {
        const int n = nb * 16 + l16;
        const float bv = b2s[n];
        #pragma unroll
        for (int r = 0; r < 8; ++r) {
            const float x  = acc[nb][r] + bv;
            const float nv = nodes[(long long)snd[r] * F + n];   // L2-resident gather
            if (rcv[r] >= 0)
                atomicAdd(out + (long long)rcv[r] * F + n, nv * x); // no-return f32 atomic
        }
    }
}

extern "C" void kernel_launch(void* const* d_in, const int* in_sizes, int n_in,
                              void* d_out, int out_size, void* d_ws, size_t ws_size,
                              hipStream_t stream) {
    const float* nodes = (const float*)d_in[0];
    const float* edges = (const float*)d_in[1];
    const int*   eidx  = (const int*)d_in[2];
    const float* W1    = (const float*)d_in[3];
    const float* b1    = (const float*)d_in[4];
    const float* W2    = (const float*)d_in[5];
    const float* b2    = (const float*)d_in[6];
    float* out = (float*)d_out;

    const int M  = in_sizes[2] / 2;   // number of edges
    const int Nn = in_sizes[0] / F;   // number of nodes

    const long long outN = (long long)Nn * F;
    cfconv_zero_kernel<<<(int)((outN + 255) / 256), 256, 0, stream>>>(out, outN);

    const int blocks = (M + BLOCK_EDGES - 1) / BLOCK_EDGES;
    cfconv_fused_kernel<<<blocks, 256, 0, stream>>>(nodes, edges, eidx,
                                                    W1, b1, W2, b2, out, M, Nn);
}
